// SpatialAttention_12120397710064
// MI455X (gfx1250) — matrix operand order repaired and checked by reading, not compiled, not run on hardware
//
#include <hip/hip_runtime.h>
#include <hip/hip_bf16.h>

// Shapes: x [B=16, TO=8, C=64, T=1024]  ->  N = 128 batches of [C=64, T=1024]
#define NB   128
#define CC   64
#define TT   1024
#define CT   (CC * TT)          // 65536 elements per batch per matrix
#define NSB  (TT / 32)          // 32 s-blocks of 32 columns

typedef __attribute__((ext_vector_type(16))) _Float16 v16h;
typedef __attribute__((ext_vector_type(8)))  _Float16 v8h;
typedef __attribute__((ext_vector_type(8)))  float    v8f;
typedef __attribute__((ext_vector_type(4)))  unsigned int u32x4;
typedef __attribute__((ext_vector_type(8)))  int      i32x8;
typedef __attribute__((ext_vector_type(4)))  int      i32x4;

__device__ __forceinline__ v8f wmma16(v16h a, v16h b, v8f c) {
    // D = A(16x32 f16) * B(32x16 f16) + C(16x16 f32)
    return __builtin_amdgcn_wmma_f32_16x16x32_f16(
        false, a, false, b, (short)0, c, false, false);
}

// A-operand (16x32, 16-bit) K-pair base for VGPR slot i, given lane half.
__device__ __forceinline__ int a_kbase(int i, int half) {
    return (half ? 16 : 0) + 2 * i;
}
// B-operand (32x16, 16-bit), this code's k order: lanes 0-15 hold K=0..15, lanes 16-31 K=16..31.
__device__ __forceinline__ int b_kbase(int i, int half) {
    return (half ? 16 : 0) + 2 * i;
}

__device__ __forceinline__ unsigned rfl(unsigned v) {
    return (unsigned)__builtin_amdgcn_readfirstlane((int)v);
}

// ---------------------------------------------------------------------------
// TDM: 2-D f16 tile DMA global -> LDS.  Descriptor per CDNA5 ISA ch.8:
//  group0: [1:0]=count=1, [63:32]=lds_addr, [120:64]=global_addr, [127:126]=2
//  group1: [17:16]=data_size(1 => 2B), [79:48]=tensor_dim0, [111:80]=tensor_dim1,
//          [127:112]=tile_dim0, [143:128]=tile_dim1, [207:160]=tensor_dim0_stride
// Tracked with TENSORcnt; completion via s_wait_tensorcnt.
// This toolchain exposes the 6-arg builtin:
//   (u32x4 g0, i32x8 g1, i32x4 g2, i32x4 g3, i32x8 extra, i32 cpol)
// ---------------------------------------------------------------------------
__device__ __forceinline__ void tdm_load_2d_f16(unsigned lds_off, const void* gp,
                                                unsigned tensor_d0, unsigned tensor_d1,
                                                unsigned tile_d0, unsigned tile_d1,
                                                unsigned stride0) {
    const unsigned long long ga = (unsigned long long)(uintptr_t)gp;
    u32x4 g0;
    g0[0] = rfl(1u);                                   // count=1 (valid user D#)
    g0[1] = rfl(lds_off);                              // lds_addr (bytes)
    g0[2] = rfl((unsigned)ga);                         // global_addr[31:0]
    g0[3] = rfl((unsigned)(ga >> 32) | (2u << 30));    // global_addr[56:32] | type=2
    i32x8 g1;
    g1[0] = (int)rfl(1u << 16);                        // data_size = 2 bytes
    g1[1] = (int)rfl((tensor_d0 & 0xFFFFu) << 16);     // tensor_dim0[15:0]
    g1[2] = (int)rfl((tensor_d0 >> 16) | ((tensor_d1 & 0xFFFFu) << 16));
    g1[3] = (int)rfl((tensor_d1 >> 16) | (tile_d0 << 16));   // tile_dim0
    g1[4] = (int)rfl(tile_d1 & 0xFFFFu);               // tile_dim1 (tile_dim2=0)
    g1[5] = (int)rfl(stride0);                         // tensor_dim0_stride[31:0]
    g1[6] = (int)rfl(0u);
    g1[7] = (int)rfl(0u);
    i32x4 z4;
    z4[0] = z4[1] = z4[2] = z4[3] = 0;                 // 2-D tensor: groups 2/3 off
    i32x8 z8;
#pragma unroll
    for (int i = 0; i < 8; ++i) z8[i] = 0;
    __builtin_amdgcn_tensor_load_to_lds(g0, g1, z4, z4, z8, 0);
}

// ---------------------------------------------------------------------------
// Kernel 1: Q/K/V projection.  One wave per (batch, t-tile): loads the x
// B-operands ONCE, then issues 3 mats x 4 o-tiles x K2 = 24 WMMAs.
//   qT (pre-scaled by 1/sqrt(C)), kT stored f16 [n][t][c]; v stored f16 [n][c][s]
// ---------------------------------------------------------------------------
__global__ void __launch_bounds__(256)
qkv_proj_kernel(const float* __restrict__ x,
                const float* __restrict__ Wq,
                const float* __restrict__ Wk,
                const float* __restrict__ Wv,
                const float* __restrict__ scale_p,
                _Float16* __restrict__ qT,
                _Float16* __restrict__ kT,
                _Float16* __restrict__ vO) {
    const int wid  = blockIdx.x * 8 + (threadIdx.x >> 5);
    const int lane = threadIdx.x & 31;
    const int half = lane >> 4;
    const int l16  = lane & 15;

    const int tIdx = wid & 63;           // 64 tiles along T
    const int nIdx = wid >> 6;           // batch 0..127
    const float scl = scale_p[0];

    // B operands from x[n][c][t]: column n-dim = t = l16, K = c (stride TT)
    const float* xb = x + (size_t)nIdx * CT + tIdx * 16 + l16;
    v16h b0, b1;
#pragma unroll
    for (int i = 0; i < 8; ++i) {
        const int k = b_kbase(i, half);
        b0[2 * i]     = (_Float16)xb[(size_t)k * TT];
        b0[2 * i + 1] = (_Float16)xb[(size_t)(k + 1) * TT];
        b1[2 * i]     = (_Float16)xb[(size_t)(32 + k) * TT];
        b1[2 * i + 1] = (_Float16)xb[(size_t)(32 + k + 1) * TT];
    }

    for (int mat = 0; mat < 3; ++mat) {
        const float* W = (mat == 0) ? Wq : (mat == 1) ? Wk : Wv;
        const float oscale = (mat == 0) ? scl : 1.0f;   // fold softmax scale into Q
#pragma unroll
        for (int oIdx = 0; oIdx < 4; ++oIdx) {
            v16h a0, a1;
            const float* wrow = W + (oIdx * 16 + l16) * CC;
#pragma unroll
            for (int i = 0; i < 8; ++i) {
                const int kb = a_kbase(i, half);
                a0[2 * i]     = (_Float16)wrow[kb];
                a0[2 * i + 1] = (_Float16)wrow[kb + 1];
                a1[2 * i]     = (_Float16)wrow[32 + kb];
                a1[2 * i + 1] = (_Float16)wrow[32 + kb + 1];
            }
            v8f acc = {};
            acc = wmma16(a0, b0, acc);
            acc = wmma16(a1, b1, acc);
            // C layout: VGPR r -> (o = oIdx*16 + r + 8*half, t = tIdx*16 + l16)
            if (mat < 2) {
                _Float16* dst = (mat == 0 ? qT : kT) + (size_t)nIdx * CT;
#pragma unroll
                for (int r = 0; r < 8; ++r) {
                    const int t = tIdx * 16 + l16;
                    const int c = oIdx * 16 + r + half * 8;
                    dst[(size_t)t * CC + c] = (_Float16)(acc[r] * oscale);
                }
            } else {
                _Float16* dst = vO + (size_t)nIdx * CT;
#pragma unroll
                for (int r = 0; r < 8; ++r) {
                    const int t = tIdx * 16 + l16;
                    const int c = oIdx * 16 + r + half * 8;
                    dst[(size_t)c * TT + t] = (_Float16)acc[r];
                }
            }
        }
    }
}

// ---------------------------------------------------------------------------
// Kernel 2: fused flash attention + output projection + residual.
// Block = 4 waves; wave w owns query rows [t0, t0+16).  K/V s-blocks are
// staged into LDS by the Tensor Data Mover (double-buffered), shared by all
// 4 waves; all GEMMs via v_wmma_f32_16x16x32_f16.
// ---------------------------------------------------------------------------
__global__ void __launch_bounds__(128)
attn_fused_kernel(const float* __restrict__ x,
                  const float* __restrict__ Wo,
                  const _Float16* __restrict__ qT,
                  const _Float16* __restrict__ kT,
                  const _Float16* __restrict__ vO,
                  float* __restrict__ out) {
    __shared__ _Float16 ldsK[2][32][CC];    // K s-block [s][c], double buffered
    __shared__ _Float16 ldsV[2][CC][32];    // V s-block [c][s], double buffered
    __shared__ _Float16 lds_p[4][16][32];   // per-wave P tile re-layout
    __shared__ _Float16 lds_o[4][16][CC];   // per-wave normalized attn tile

    const int w    = threadIdx.x >> 5;
    const int lane = threadIdx.x & 31;
    const int half = lane >> 4;
    const int l16  = lane & 15;

    const int nb   = blockIdx.x >> 4;
    const int qblk = blockIdx.x & 15;
    const int t0   = qblk * 64 + w * 16;

    const _Float16* qbase = qT + (size_t)nb * CT;   // [t][c] (pre-scaled)
    const _Float16* kbase = kT + (size_t)nb * CT;   // [s][c]
    const _Float16* vbase = vO + (size_t)nb * CT;   // [c][s]

    // Q as A-operand: (m = t-row = l16, k = c); two contiguous 8-elem chunks.
    v16h qa0, qa1;
    {
        const _Float16* qr = qbase + (size_t)(t0 + l16) * CC;
        const int kb = half ? 16 : 0;
        v8h q00 = *(const v8h*)(qr + kb);
        v8h q01 = *(const v8h*)(qr + 8 + kb);
        v8h q10 = *(const v8h*)(qr + 32 + kb);
        v8h q11 = *(const v8h*)(qr + 40 + kb);
#pragma unroll
        for (int i = 0; i < 8; ++i) {
            qa0[i] = q00[i]; qa0[8 + i] = q01[i];
            qa1[i] = q10[i]; qa1[8 + i] = q11[i];
        }
    }

    v8f o0 = {}, o1 = {}, o2 = {}, o3 = {};
    float mrow[8], lrow[8];
#pragma unroll
    for (int r = 0; r < 8; ++r) { mrow[r] = -1e30f; lrow[r] = 0.0f; }

    // Prologue: TDM-stage s-block 0 into buffer 0 (wave 0 only; EXEC ignored).
    if (w == 0) {
        tdm_load_2d_f16((unsigned)(uintptr_t)&ldsK[0][0][0],
                        kbase, CC, TT, CC, 32, CC);
        tdm_load_2d_f16((unsigned)(uintptr_t)&ldsV[0][0][0],
                        vbase, TT, CC, 32, CC, TT);
    }

    for (int sb = 0; sb < NSB; ++sb) {
        const int buf = sb & 1;
        if (w == 0) {
            if (sb + 1 < NSB) {   // kick next block, then wait for current
                const int nxt = buf ^ 1;
                tdm_load_2d_f16((unsigned)(uintptr_t)&ldsK[nxt][0][0],
                                kbase + (size_t)(sb + 1) * 32 * CC,
                                CC, TT, CC, 32, CC);
                tdm_load_2d_f16((unsigned)(uintptr_t)&ldsV[nxt][0][0],
                                vbase + (sb + 1) * 32,
                                TT, CC, 32, CC, TT);
                __builtin_amdgcn_s_wait_tensorcnt((short)2);
            } else {
                __builtin_amdgcn_s_wait_tensorcnt((short)0);
            }
        }
        __syncthreads();   // current K/V s-block resident in LDS

        // ---- scores: S = Q^T K, two 16-wide s sub-tiles, K=c=64 ----
        v8f s0 = {}, s1 = {};
#pragma unroll
        for (int j = 0; j < 2; ++j) {
            const _Float16* kr = &ldsK[buf][j * 16 + l16][0];
            const int kb = half ? 16 : 0;
            v16h b0 = *(const v16h*)(kr + kb);        // c 0..31 slice
            v16h b1 = *(const v16h*)(kr + 32 + kb);   // c 32..63 slice
            v8f sc = {};
            sc = wmma16(qa0, b0, sc);
            sc = wmma16(qa1, b1, sc);
            if (j == 0) s0 = sc; else s1 = sc;
        }

        // ---- online softmax over these 32 columns ----
#pragma unroll
        for (int r = 0; r < 8; ++r) {
            float a = s0[r], b = s1[r];
            float rm = fmaxf(a, b);
#pragma unroll
            for (int msk = 8; msk >= 1; msk >>= 1)
                rm = fmaxf(rm, __shfl_xor(rm, msk, 16));
            const float mn   = fmaxf(mrow[r], rm);
            const float corr = __expf(mrow[r] - mn);
            const float p0   = __expf(a - mn);
            const float p1   = __expf(b - mn);
            float rs = p0 + p1;
#pragma unroll
            for (int msk = 8; msk >= 1; msk >>= 1)
                rs += __shfl_xor(rs, msk, 16);
            lrow[r] = lrow[r] * corr + rs;
            mrow[r] = mn;
            o0[r] *= corr; o1[r] *= corr; o2[r] *= corr; o3[r] *= corr;
            const int trow = r + half * 8;
            lds_p[w][trow][l16]      = (_Float16)p0;
            lds_p[w][trow][16 + l16] = (_Float16)p1;
        }
        __syncthreads();   // order per-wave P writes vs A-layout reads

        // ---- P as A-operand (m = t, k = s local 0..31) ----
        v16h pa;
        {
            const _Float16* pr = &lds_p[w][l16][0];
            const int kb = half ? 16 : 0;
            v8h plo = *(const v8h*)(pr + kb);
            v8h phi = *(const v8h*)(pr + 8 + kb);
#pragma unroll
            for (int i = 0; i < 8; ++i) { pa[i] = plo[i]; pa[8 + i] = phi[i]; }
        }

        // ---- O += P · V^T : B(k=s, n=c) from ldsV[c][s] ----
#pragma unroll
        for (int cc = 0; cc < 4; ++cc) {
            const _Float16* vr = &ldsV[buf][cc * 16 + l16][0];
            v16h b = *(const v16h*)(vr + (half ? 16 : 0));
            if      (cc == 0) o0 = wmma16(pa, b, o0);
            else if (cc == 1) o1 = wmma16(pa, b, o1);
            else if (cc == 2) o2 = wmma16(pa, b, o2);
            else              o3 = wmma16(pa, b, o3);
        }
        __syncthreads();   // all waves done with buf before TDM reuses it
    }

    // ---- normalize and stage attention tile [t][c] in LDS ----
#pragma unroll
    for (int r = 0; r < 8; ++r) {
        const float inv = 1.0f / lrow[r];
        const int trow = r + half * 8;
        lds_o[w][trow][l16]      = (_Float16)(o0[r] * inv);
        lds_o[w][trow][16 + l16] = (_Float16)(o1[r] * inv);
        lds_o[w][trow][32 + l16] = (_Float16)(o2[r] * inv);
        lds_o[w][trow][48 + l16] = (_Float16)(o3[r] * inv);
    }
    __syncthreads();

    // ---- output projection: final = Wo @ attn^T + x ----
#pragma unroll
    for (int oo = 0; oo < 4; ++oo) {
        v16h a0, a1;
        const float* wrow = Wo + (oo * 16 + l16) * CC;
#pragma unroll
        for (int i = 0; i < 8; ++i) {
            const int kb = a_kbase(i, half);
            a0[2 * i]     = (_Float16)wrow[kb];
            a0[2 * i + 1] = (_Float16)wrow[kb + 1];
            a1[2 * i]     = (_Float16)wrow[32 + kb];
            a1[2 * i + 1] = (_Float16)wrow[32 + kb + 1];
        }
        const _Float16* orow = &lds_o[w][l16][0];
        const int kb = half ? 16 : 0;
        v16h b0 = *(const v16h*)(orow + kb);
        v16h b1 = *(const v16h*)(orow + 32 + kb);
        v8f acc = {};
        acc = wmma16(a0, b0, acc);
        acc = wmma16(a1, b1, acc);
        // D layout: (o = oo*16 + r + 8*half, t = t0 + l16); add residual x
#pragma unroll
        for (int r = 0; r < 8; ++r) {
            const size_t idx = (size_t)nb * CT +
                               (size_t)(oo * 16 + r + half * 8) * TT +
                               (size_t)(t0 + l16);
            out[idx] = acc[r] + x[idx];
        }
    }
}

extern "C" void kernel_launch(void* const* d_in, const int* in_sizes, int n_in,
                              void* d_out, int out_size, void* d_ws, size_t ws_size,
                              hipStream_t stream) {
    const float* x     = (const float*)d_in[0];
    const float* Wq    = (const float*)d_in[1];
    const float* Wk    = (const float*)d_in[2];
    const float* Wv    = (const float*)d_in[3];
    const float* Wo    = (const float*)d_in[4];
    const float* scale = (const float*)d_in[5];

    _Float16* ws = (_Float16*)d_ws;
    _Float16* qT = ws;                            // [128][1024][64] f16
    _Float16* kT = ws + (size_t)NB * CT;          // [128][1024][64] f16
    _Float16* vO = ws + (size_t)2 * NB * CT;      // [128][64][1024] f16

    // 128 batches * 64 t-tiles = 8192 waves / 8 per block
    qkv_proj_kernel<<<1024, 256, 0, stream>>>(x, Wq, Wk, Wv, scale, qT, kT, vO);
    // 128 batches * 16 query blocks of 64 rows
    attn_fused_kernel<<<2048, 128, 0, stream>>>(x, Wo, qT, kT, vO,
                                                (float*)d_out);
}
